// BinaryLabelSoftRouter_34084860461659
// MI455X (gfx1250) — matrix operand order repaired and checked by reading, not compiled, not run on hardware
//
#include <hip/hip_runtime.h>
#include <hip/hip_bf16.h>
#include <cstddef>

typedef __attribute__((ext_vector_type(16))) _Float16 v16h;
typedef __attribute__((ext_vector_type(8)))  _Float16 h8;
typedef __attribute__((ext_vector_type(8)))  float    v8f;

#define TOKENS 16384
#define K1 2048
#define N1 256
#define N2 128
#define LN_EPS 1e-5f
#define GELU_K 0.70710678118654752f

__device__ __forceinline__ float gelu_exact(float x) {
    return 0.5f * x * (1.0f + erff(x * GELU_K));
}

// ---------------------------------------------------------------------------
// Weight pre-pass: W[K,N] f32  ->  WT[N,K] f16 (LDS-tiled transpose, 32x32)
// ---------------------------------------------------------------------------
__global__ __launch_bounds__(256) void transpose_to_f16(
    const float* __restrict__ W, _Float16* __restrict__ WT, int K, int N)
{
    __shared__ _Float16 tile[32][33];
    const int kb = blockIdx.x * 32, nb = blockIdx.y * 32;
    const int c  = threadIdx.x & 31;
    const int r4 = threadIdx.x >> 5;          // 0..7
    #pragma unroll
    for (int i = 0; i < 4; ++i) {
        int r = r4 * 4 + i;
        tile[r][c] = (_Float16)W[(size_t)(kb + r) * N + nb + c];
    }
    __syncthreads();
    #pragma unroll
    for (int i = 0; i < 4; ++i) {
        int r = r4 * 4 + i;                   // n-offset
        WT[(size_t)(nb + r) * K + kb + c] = tile[c][r];
    }
}

// ---------------------------------------------------------------------------
// Phase 1: h1 = gelu(LN(X @ W1 + b1))
// X:[16384,2048] f32, W1T:[256,2048] f16 -> H1:[16384,256] f16
// 256 blocks x 256 threads (8 waves: 4 M-tiles x 2 N-halves), 64 rows/block.
// In-register LN epilogue; only 1KB of per-row partials cross LDS.
// ---------------------------------------------------------------------------
__global__ __launch_bounds__(256) void gemm1_ln_gelu(
    const float* __restrict__ X, const _Float16* __restrict__ W1T,
    const float* __restrict__ b1, const float* __restrict__ g1,
    const float* __restrict__ be1, _Float16* __restrict__ H1)
{
    __shared__ char smem[30720];
    _Float16* Xs = (_Float16*)smem;            // 64 rows * 40 halves (80B padded)
    _Float16* Ws = (_Float16*)(smem + 5120);   // 256 cols * 40 halves
    float*    pS = (float*)smem;               // reused: 64 rows * 2 halves
    float*    pQ = (float*)smem + 128;

    const int tid   = threadIdx.x;
    const int wv    = tid >> 5;
    const int lane  = tid & 31;
    const int mwave = wv & 3;        // 4 M-tiles of 16 rows
    const int nwave = wv >> 2;       // 2 N-halves of 128 cols
    const int rl    = lane & 15;
    const int hi    = lane >> 4;
    const int blockRow = blockIdx.x * 64;

    v8f acc[8];
    #pragma unroll
    for (int i = 0; i < 8; ++i) acc[i] = (v8f){0.f,0.f,0.f,0.f,0.f,0.f,0.f,0.f};

    const int xr = tid >> 2;          // 0..63 row within block
    const int xj = (tid & 3) * 8;     // 0,8,16,24 k-offset

    for (int kc = 0; kc < K1; kc += 32) {
        __syncthreads();
        // ---- stage X tile (64 x 32) f32 -> f16
        {
            const float* src = X + (size_t)(blockRow + xr) * K1 + kc + xj;
            float4 f0 = *(const float4*)src;
            float4 f1 = *(const float4*)(src + 4);
            h8 hh;
            hh[0]=(_Float16)f0.x; hh[1]=(_Float16)f0.y; hh[2]=(_Float16)f0.z; hh[3]=(_Float16)f0.w;
            hh[4]=(_Float16)f1.x; hh[5]=(_Float16)f1.y; hh[6]=(_Float16)f1.z; hh[7]=(_Float16)f1.w;
            *(h8*)(Xs + xr * 40 + xj) = hh;
        }
        // ---- stage W1T chunk: pure b128 copies, col n = tid, 32 k-halves
        {
            const _Float16* wsrc = W1T + (size_t)tid * K1 + kc;
            #pragma unroll
            for (int g = 0; g < 4; ++g)
                *(h8*)(Ws + tid * 40 + g * 8) = *(const h8*)(wsrc + g * 8);
        }
        __syncthreads();
        // ---- A fragment: lane<16 -> row rl, K{0..7,16..23}; lane>=16 -> K{8..15,24..31}
        v16h A;
        const _Float16* aBase = Xs + (mwave * 16 + rl) * 40;
        ((h8*)&A)[0] = *(const h8*)(aBase + hi * 8);
        ((h8*)&A)[1] = *(const h8*)(aBase + 16 + hi * 8);
        #pragma unroll
        for (int nt = 0; nt < 8; ++nt) {
            const _Float16* bBase = Ws + (nwave * 128 + nt * 16 + rl) * 40 + hi * 8;
            v16h Bv;
            ((h8*)&Bv)[0] = *(const h8*)(bBase);
            ((h8*)&Bv)[1] = *(const h8*)(bBase + 16);
            acc[nt] = __builtin_amdgcn_wmma_f32_16x16x32_f16(
                false, A, false, Bv, (short)0, acc[nt], false, false);
        }
    }
    __syncthreads();   // LDS now reusable for partials

    // ---- bias add (in registers, before LN statistics)
    const int colBase = nwave * 128 + rl;
    #pragma unroll
    for (int nt = 0; nt < 8; ++nt) {
        float b = b1[colBase + nt * 16];
        #pragma unroll
        for (int v = 0; v < 8; ++v) acc[nt][v] += b;
    }
    // ---- per-row partial sums: local over nt, half-wave reduce over 16 lanes
    #pragma unroll
    for (int v = 0; v < 8; ++v) {
        float s = 0.f, q = 0.f;
        #pragma unroll
        for (int nt = 0; nt < 8; ++nt) { float y = acc[nt][v]; s += y; q += y * y; }
        #pragma unroll
        for (int off = 8; off >= 1; off >>= 1) {
            s += __shfl_xor(s, off, 32);
            q += __shfl_xor(q, off, 32);
        }
        if (rl == 0) {
            int row = mwave * 16 + v + 8 * hi;
            pS[row * 2 + nwave] = s;
            pQ[row * 2 + nwave] = q;
        }
    }
    __syncthreads();
    // ---- normalize + gelu straight from accumulators
    float gam[8], bet[8];
    #pragma unroll
    for (int nt = 0; nt < 8; ++nt) {
        gam[nt] = g1[colBase + nt * 16];
        bet[nt] = be1[colBase + nt * 16];
    }
    #pragma unroll
    for (int v = 0; v < 8; ++v) {
        int row = mwave * 16 + v + 8 * hi;
        float s  = pS[row * 2] + pS[row * 2 + 1];
        float q  = pQ[row * 2] + pQ[row * 2 + 1];
        float mu = s * (1.f / 256.f);
        float var = q * (1.f / 256.f) - mu * mu;
        float rq = rsqrtf(var + LN_EPS);
        _Float16* dst = H1 + (size_t)(blockRow + row) * 256 + colBase;
        #pragma unroll
        for (int nt = 0; nt < 8; ++nt) {
            float u = (acc[nt][v] - mu) * rq * gam[nt] + bet[nt];
            dst[nt * 16] = (_Float16)gelu_exact(u);
        }
    }
}

// ---------------------------------------------------------------------------
// Phase 2: h2 = gelu(LN(h1 @ W2 + b2))
// H1:[16384,256] f16, W2T:[128,256] f16 -> H2:[16384,128] f16
// ---------------------------------------------------------------------------
__global__ __launch_bounds__(256) void gemm2_ln_gelu(
    const _Float16* __restrict__ H1, const _Float16* __restrict__ W2T,
    const float* __restrict__ b2, const float* __restrict__ g2,
    const float* __restrict__ be2, _Float16* __restrict__ H2)
{
    __shared__ char smem[15360];
    _Float16* Xs = (_Float16*)smem;            // 64 * 40 halves
    _Float16* Ws = (_Float16*)(smem + 5120);   // 128 * 40 halves
    float*    pS = (float*)smem;
    float*    pQ = (float*)smem + 128;

    const int tid   = threadIdx.x;
    const int wv    = tid >> 5;
    const int lane  = tid & 31;
    const int mwave = wv & 3;
    const int nwave = wv >> 2;       // 2 N-halves of 64 cols
    const int rl    = lane & 15;
    const int hi    = lane >> 4;
    const int blockRow = blockIdx.x * 64;

    v8f acc[4];
    #pragma unroll
    for (int i = 0; i < 4; ++i) acc[i] = (v8f){0.f,0.f,0.f,0.f,0.f,0.f,0.f,0.f};

    const int xr = tid >> 2;
    const int xj = (tid & 3) * 8;
    const int wn = tid >> 1;          // 0..127 col
    const int wj = (tid & 1) * 16;    // k-half

    for (int kc = 0; kc < N1; kc += 32) {
        __syncthreads();
        *(h8*)(Xs + xr * 40 + xj) =
            *(const h8*)(H1 + (size_t)(blockRow + xr) * 256 + kc + xj);
        {
            const _Float16* wsrc = W2T + (size_t)wn * N1 + kc + wj;
            *(h8*)(Ws + wn * 40 + wj)     = *(const h8*)(wsrc);
            *(h8*)(Ws + wn * 40 + wj + 8) = *(const h8*)(wsrc + 8);
        }
        __syncthreads();
        v16h A;
        const _Float16* aBase = Xs + (mwave * 16 + rl) * 40;
        ((h8*)&A)[0] = *(const h8*)(aBase + hi * 8);
        ((h8*)&A)[1] = *(const h8*)(aBase + 16 + hi * 8);
        #pragma unroll
        for (int nt = 0; nt < 4; ++nt) {
            const _Float16* bBase = Ws + (nwave * 64 + nt * 16 + rl) * 40 + hi * 8;
            v16h Bv;
            ((h8*)&Bv)[0] = *(const h8*)(bBase);
            ((h8*)&Bv)[1] = *(const h8*)(bBase + 16);
            acc[nt] = __builtin_amdgcn_wmma_f32_16x16x32_f16(
                false, A, false, Bv, (short)0, acc[nt], false, false);
        }
    }
    __syncthreads();

    const int colBase = nwave * 64 + rl;
    #pragma unroll
    for (int nt = 0; nt < 4; ++nt) {
        float b = b2[colBase + nt * 16];
        #pragma unroll
        for (int v = 0; v < 8; ++v) acc[nt][v] += b;
    }
    #pragma unroll
    for (int v = 0; v < 8; ++v) {
        float s = 0.f, q = 0.f;
        #pragma unroll
        for (int nt = 0; nt < 4; ++nt) { float y = acc[nt][v]; s += y; q += y * y; }
        #pragma unroll
        for (int off = 8; off >= 1; off >>= 1) {
            s += __shfl_xor(s, off, 32);
            q += __shfl_xor(q, off, 32);
        }
        if (rl == 0) {
            int row = mwave * 16 + v + 8 * hi;
            pS[row * 2 + nwave] = s;
            pQ[row * 2 + nwave] = q;
        }
    }
    __syncthreads();
    float gam[4], bet[4];
    #pragma unroll
    for (int nt = 0; nt < 4; ++nt) {
        gam[nt] = g2[colBase + nt * 16];
        bet[nt] = be2[colBase + nt * 16];
    }
    #pragma unroll
    for (int v = 0; v < 8; ++v) {
        int row = mwave * 16 + v + 8 * hi;
        float s  = pS[row * 2] + pS[row * 2 + 1];
        float q  = pQ[row * 2] + pQ[row * 2 + 1];
        float mu = s * (1.f / 128.f);
        float var = q * (1.f / 128.f) - mu * mu;
        float rq = rsqrtf(var + LN_EPS);
        _Float16* dst = H2 + (size_t)(blockRow + row) * 128 + colBase;
        #pragma unroll
        for (int nt = 0; nt < 4; ++nt) {
            float u = (acc[nt][v] - mu) * rq * gam[nt] + bet[nt];
            dst[nt * 16] = (_Float16)gelu_exact(u);
        }
    }
}

// ---------------------------------------------------------------------------
// Phase 3: adjusted = softmax((base + tanh(h2@W3 + b3)*0.1)/clip(temp,0.1))
// ---------------------------------------------------------------------------
__global__ __launch_bounds__(256) void head_softmax(
    const _Float16* __restrict__ H2, const float* __restrict__ W3,
    const float* __restrict__ b3, const int* __restrict__ labels,
    const float* __restrict__ tempPtr, float* __restrict__ Adj)
{
    int tok = blockIdx.x * 256 + threadIdx.x;
    if (tok >= TOKENS) return;
    float a0 = b3[0], a1 = b3[1];
    const _Float16* h = H2 + (size_t)tok * 128;
    #pragma unroll 8
    for (int k = 0; k < 128; ++k) {
        float v = (float)h[k];
        a0 += v * W3[k * 2 + 0];
        a1 += v * W3[k * 2 + 1];
    }
    a0 = tanhf(a0) * 0.1f;
    a1 = tanhf(a1) * 0.1f;
    int lab = labels[tok];
    float base0 = (lab > 0) ? 0.25f : 0.75f;
    float base1 = (lab > 0) ? 0.75f : 0.25f;
    float itemp = 1.0f / fmaxf(tempPtr[0], 0.1f);
    float l0 = (base0 + a0) * itemp, l1 = (base1 + a1) * itemp;
    float m  = fmaxf(l0, l1);
    float e0 = expf(l0 - m), e1 = expf(l1 - m);
    float inv = 1.0f / (e0 + e1);
    Adj[tok * 2 + 0] = e0 * inv;
    Adj[tok * 2 + 1] = e1 * inv;
}

// ---------------------------------------------------------------------------
// Phase 4: EMA scan, windowed closed form (0.9^512 ~ 4e-24, exact in f32).
// ---------------------------------------------------------------------------
__global__ __launch_bounds__(256) void ema_smooth(
    const float* __restrict__ Adj, float* __restrict__ Out)
{
    __shared__ float2 c[1024];
    const int b = blockIdx.x;
    const float2* src = (const float2*)(Adj + (size_t)b * 2048);
    for (int i = threadIdx.x; i < 1024; i += 256) c[i] = src[i];
    __syncthreads();
    float2* dst = (float2*)(Out + (size_t)b * 2048);
    for (int t = threadIdx.x; t < 1024; t += 256) {
        float2 r;
        if (t == 0) {
            r = c[0];
        } else {
            float s0 = 0.f, s1 = 0.f, f = 0.1f;
            int kmin = t - 511; if (kmin < 1) kmin = 1;
            for (int k = t; k >= kmin; --k) {
                float2 v = c[k];
                s0 += f * v.x; s1 += f * v.y;
                f *= 0.9f;
            }
            if (kmin == 1) {              // c0 coefficient is 0.9^t = f*10
                float2 v = c[0];
                float f0 = f * 10.0f;
                s0 += f0 * v.x; s1 += f0 * v.y;
            }
            r.x = s0; r.y = s1;
        }
        dst[t] = r;
    }
}

extern "C" void kernel_launch(void* const* d_in, const int* in_sizes, int n_in,
                              void* d_out, int out_size, void* d_ws, size_t ws_size,
                              hipStream_t stream) {
    const int*   labels = (const int*)  d_in[0];
    const float* X      = (const float*)d_in[1];
    const float* W1     = (const float*)d_in[2];
    const float* b1     = (const float*)d_in[3];
    const float* g1     = (const float*)d_in[4];
    const float* be1    = (const float*)d_in[5];
    const float* W2     = (const float*)d_in[6];
    const float* b2     = (const float*)d_in[7];
    const float* g2     = (const float*)d_in[8];
    const float* be2    = (const float*)d_in[9];
    const float* W3     = (const float*)d_in[10];
    const float* b3     = (const float*)d_in[11];
    const float* temp   = (const float*)d_in[12];

    char* ws = (char*)d_ws;
    _Float16* H1  = (_Float16*)(ws);                                  // 8 MB
    _Float16* H2  = (_Float16*)(ws + (size_t)8  * 1024 * 1024);       // 4 MB
    float*    Adj = (float*)   (ws + (size_t)12 * 1024 * 1024);       // 128 KB
    _Float16* W1T = (_Float16*)(ws + (size_t)12 * 1024 * 1024 + 512 * 1024); // 1 MB
    _Float16* W2T = (_Float16*)(ws + (size_t)13 * 1024 * 1024 + 512 * 1024); // 64 KB
    float*    Out = (float*)d_out;

    transpose_to_f16<<<dim3(K1 / 32, N1 / 32), 256, 0, stream>>>(W1, W1T, K1, N1);
    transpose_to_f16<<<dim3(N1 / 32, N2 / 32), 256, 0, stream>>>(W2, W2T, N1, N2);
    gemm1_ln_gelu<<<TOKENS / 64, 256, 0, stream>>>(X, W1T, b1, g1, be1, H1);
    gemm2_ln_gelu<<<TOKENS / 64, 256, 0, stream>>>(H1, W2T, b2, g2, be2, H2);
    head_softmax <<<TOKENS / 256, 256, 0, stream>>>(H2, W3, b3, labels, temp, Adj);
    ema_smooth   <<<16, 256, 0, stream>>>(Adj, Out);
}
